// RGCN_53901839565613
// MI455X (gfx1250) — matrix operand order repaired and checked by reading, not compiled, not run on hardware
//
#include <hip/hip_runtime.h>
#include <hip/hip_bf16.h>

// ---------------------------------------------------------------------------
// RGCN (2 layers) for MI455X / gfx1250, wave32 + WMMA bf16.
//
// Rewrite: agg[n] = sum_b q[n,b] @ V_b,  q[n,b] = sum_r comp[r,b]*pre[n,r],
//          pre[n,r] = scatter-sum of raw h[src] over edges with dst=n,etype=r.
// Then out = [h | q0..q3] (N x 640, bf16) @ [loopW; V0..V3] (640 x 128, bf16)
//          + bias (+ ReLU), computed with v_wmma_f32_16x16x32_bf16.
// ---------------------------------------------------------------------------

typedef __bf16 bf16_t;
typedef __attribute__((ext_vector_type(16))) __bf16 v16bf;
typedef __attribute__((ext_vector_type(8)))  float  v8f;

#define N_NODES 100000
#define N_EDGES 600000
#define HDIM    128
#define NREL    8
#define NBASE   4
#define KTOT    640                 // (NBASE+1)*HDIM
#define NPAD    100096              // 782 * 128 (GEMM M padding)
#define KSTEPS  20                  // KTOT / 32
#define WFRAG_ELEMS (KSTEPS*8*32*16)     // 81920 bf16
#define WFRAG_BYTES (WFRAG_ELEMS*2)      // 163840 B (fits in 320KB WGP LDS)

#define PRE_BYTES ((size_t)N_NODES * NREL * HDIM * 4)   // 409.6 MB
#define X_BYTES   ((size_t)NPAD * KTOT * 2)             // 128.1 MB
#define H1_BYTES  ((size_t)N_NODES * HDIM * 4)          // 51.2 MB

// ---------------------------------------------------------------------------
// Kernel 1: per-edge scatter of raw features into pre[dst, etype, :].
// One wave (32 lanes) per edge; each lane moves 4 contiguous floats
// (float4 load) and issues 4 global_atomic_add_f32.
// ---------------------------------------------------------------------------
__global__ __launch_bounds__(256) void edge_scatter(
    const float* __restrict__ h, const int* __restrict__ src,
    const int* __restrict__ dst, const int* __restrict__ ety,
    float* __restrict__ pre)
{
    int e    = (blockIdx.x * 256 + threadIdx.x) >> 5;   // E divisible by 8
    int lane = threadIdx.x & 31;
    int s = src[e], d = dst[e], r = ety[e];
    const float4 v = ((const float4*)(h + (size_t)s * HDIM))[lane];
    float* p = pre + ((size_t)d * NREL + r) * HDIM + lane * 4;
    __hip_atomic_fetch_add(p + 0, v.x, __ATOMIC_RELAXED, __HIP_MEMORY_SCOPE_AGENT);
    __hip_atomic_fetch_add(p + 1, v.y, __ATOMIC_RELAXED, __HIP_MEMORY_SCOPE_AGENT);
    __hip_atomic_fetch_add(p + 2, v.z, __ATOMIC_RELAXED, __HIP_MEMORY_SCOPE_AGENT);
    __hip_atomic_fetch_add(p + 3, v.w, __ATOMIC_RELAXED, __HIP_MEMORY_SCOPE_AGENT);
}

// ---------------------------------------------------------------------------
// Kernel 2: fold comp into pre -> q, pack X = [h | q0..q3] in bf16.
// One thread per (node, column); coalesced across columns.
// ---------------------------------------------------------------------------
__global__ __launch_bounds__(256) void build_x(
    const float* __restrict__ h, const float* __restrict__ pre,
    const float* __restrict__ comp, bf16_t* __restrict__ X)
{
    int gid = blockIdx.x * 256 + threadIdx.x;   // N*H threads exactly
    int n = gid >> 7;
    int c = gid & 127;
    float q[NBASE] = {0.f, 0.f, 0.f, 0.f};
    const float* pr = pre + (size_t)n * NREL * HDIM + c;
#pragma unroll
    for (int r = 0; r < NREL; ++r) {
        float pv = pr[(size_t)r * HDIM];
#pragma unroll
        for (int b = 0; b < NBASE; ++b) q[b] += comp[r * NBASE + b] * pv;
    }
    bf16_t* xr = X + (size_t)n * KTOT;
    xr[c] = (bf16_t)h[(size_t)n * HDIM + c];
#pragma unroll
    for (int b = 0; b < NBASE; ++b) xr[HDIM * (b + 1) + c] = (bf16_t)q[b];
}

// ---------------------------------------------------------------------------
// Kernel 3: build the 640x128 weight [loopW; V0..V3] in bf16, pre-swizzled
// into v_wmma_f32_16x16x32_bf16 B-fragment order:
//   fid = ((kk*8 + nsub)*32 + lane)*16 + j
//   K = kk*32 + (j&7) + 8*(lane>=16) + 16*(j>>3),  N = nsub*16 + (lane&15)
// ---------------------------------------------------------------------------
__global__ __launch_bounds__(256) void prep_w(
    const float* __restrict__ loopW, const float* __restrict__ V,
    bf16_t* __restrict__ Wfrag)
{
    int fid = blockIdx.x * 256 + threadIdx.x;   // WFRAG_ELEMS exactly
    int j    = fid & 15;
    int lane = (fid >> 4) & 31;
    int nsub = (fid >> 9) & 7;
    int kk   = fid >> 12;
    int k = kk * 32 + (j & 7) + ((lane >> 4) << 3) + ((j >> 3) << 4);
    int n = nsub * 16 + (lane & 15);
    float w;
    if (k < HDIM) {
        w = loopW[k * HDIM + n];
    } else {
        int b  = (k - HDIM) >> 7;
        int ki = k & 127;
        w = V[((size_t)b * HDIM + ki) * HDIM + n];
    }
    Wfrag[fid] = (bf16_t)w;
}

// ---------------------------------------------------------------------------
// Kernel 4: GEMM out[M,128] = X[M,640] @ W[640,128] + bias (+ReLU), bf16 WMMA.
// Block = 256 threads = 8 waves; 128x128 output tile per block.
// Wave w owns rows [w*16, w*16+16) x all 128 cols -> 8 accumulators (v8f).
// Whole pre-swizzled weight (160 KB) staged once in LDS per block.
// A fragment per the 16-bit A layout: lane&15 = M row; lane>>4 selects
// K {0..7,16..23} vs {8..15,24..31} -> two contiguous 16B loads per lane.
// ---------------------------------------------------------------------------
__global__ __launch_bounds__(256) void gemm_wmma(
    const bf16_t* __restrict__ X, const bf16_t* __restrict__ Wfrag,
    const float* __restrict__ bias, float* __restrict__ out, int relu)
{
    extern __shared__ __align__(16) char lds_raw[];
    uint4*       ldsv = (uint4*)lds_raw;
    const uint4* wv   = (const uint4*)Wfrag;
    for (int i = threadIdx.x; i < WFRAG_BYTES / 16; i += 256) ldsv[i] = wv[i];
    __syncthreads();

    const int wave   = threadIdx.x >> 5;
    const int lane   = threadIdx.x & 31;
    const int hi     = lane >> 4;
    const int lanelo = lane & 15;
    const int m_tile = blockIdx.x * 128 + wave * 16;

    const bf16_t* arow = X + (size_t)(m_tile + lanelo) * KTOT;

    v8f acc[8];
#pragma unroll
    for (int ns = 0; ns < 8; ++ns) acc[ns] = (v8f)0.f;

    for (int kk = 0; kk < KSTEPS; ++kk) {
        union { uint4 u[2]; v16bf v; } A;
        const bf16_t* ap = arow + kk * 32 + hi * 8;
        A.u[0] = *(const uint4*)(ap);        // K: hi*8 .. hi*8+7
        A.u[1] = *(const uint4*)(ap + 16);   // K: 16+hi*8 .. 16+hi*8+7
#pragma unroll
        for (int ns = 0; ns < 8; ++ns) {
            union { uint4 u[2]; v16bf v; } B;
            int bidx = ((kk * 8 + ns) * 32 + lane) * 2;  // uint4 units
            B.u[0] = ldsv[bidx];
            B.u[1] = ldsv[bidx + 1];
            acc[ns] = __builtin_amdgcn_wmma_f32_16x16x32_bf16(
                false, A.v, false, B.v, (short)0, acc[ns], false, false);
        }
    }

    // Epilogue: C/D layout -> row = m_tile + v + 8*hi, col = ns*16 + lanelo.
#pragma unroll
    for (int ns = 0; ns < 8; ++ns) {
        int col = ns * 16 + lanelo;
        float bcol = bias[col];
#pragma unroll
        for (int v = 0; v < 8; ++v) {
            int row = m_tile + v + 8 * hi;
            if (row < N_NODES) {
                float val = acc[ns][v] + bcol;
                if (relu) val = fmaxf(val, 0.f);
                out[(size_t)row * HDIM + col] = val;
            }
        }
    }
}

// ---------------------------------------------------------------------------
// Launcher. Inputs (setup_inputs order):
// 0 node_ids (identity, unused), 1 src, 2 dst, 3 etype, 4 embedding,
// 5 V1, 6 comp1, 7 loop1, 8 bias1, 9 V2, 10 comp2, 11 loop2, 12 bias2.
// ---------------------------------------------------------------------------
extern "C" void kernel_launch(void* const* d_in, const int* in_sizes, int n_in,
                              void* d_out, int out_size, void* d_ws, size_t ws_size,
                              hipStream_t stream)
{
    const int*   src   = (const int*)d_in[1];
    const int*   dst   = (const int*)d_in[2];
    const int*   ety   = (const int*)d_in[3];
    const float* emb   = (const float*)d_in[4];
    const float* V1    = (const float*)d_in[5];
    const float* comp1 = (const float*)d_in[6];
    const float* loop1 = (const float*)d_in[7];
    const float* bias1 = (const float*)d_in[8];
    const float* V2    = (const float*)d_in[9];
    const float* comp2 = (const float*)d_in[10];
    const float* loop2 = (const float*)d_in[11];
    const float* bias2 = (const float*)d_in[12];

    char*   ws   = (char*)d_ws;
    float*  pre  = (float*)(ws);
    bf16_t* X    = (bf16_t*)(ws + PRE_BYTES);
    bf16_t* Wf   = (bf16_t*)(ws + PRE_BYTES + X_BYTES);
    float*  h1   = (float*)(ws + PRE_BYTES + X_BYTES + ((WFRAG_BYTES + 255) & ~255));
    float*  outp = (float*)d_out;

    const int eb = (N_EDGES * 32) / 256;        // 75000 blocks
    const int xb = (N_NODES * HDIM) / 256;      // 50000 blocks
    const int wb = WFRAG_ELEMS / 256;           // 320 blocks
    const int gb = NPAD / 128;                  // 782 blocks

    // ---- layer 1: emb -> h1 (ReLU) ----
    hipMemsetAsync(pre, 0, PRE_BYTES, stream);
    edge_scatter<<<eb, 256, 0, stream>>>(emb, src, dst, ety, pre);
    build_x<<<xb, 256, 0, stream>>>(emb, pre, comp1, X);
    prep_w<<<wb, 256, 0, stream>>>(loop1, V1, Wf);
    gemm_wmma<<<gb, 256, WFRAG_BYTES, stream>>>(X, Wf, bias1, h1, 1);

    // ---- layer 2: h1 -> out (no activation) ----
    hipMemsetAsync(pre, 0, PRE_BYTES, stream);
    edge_scatter<<<eb, 256, 0, stream>>>(h1, src, dst, ety, pre);
    build_x<<<xb, 256, 0, stream>>>(h1, pre, comp2, X);
    prep_w<<<wb, 256, 0, stream>>>(loop2, V2, Wf);
    gemm_wmma<<<gb, 256, WFRAG_BYTES, stream>>>(X, Wf, bias2, outp, 0);
}
